// ESLSTMCell_21904333209947
// MI455X (gfx1250) — matrix operand order repaired and checked by reading, not compiled, not run on hardware
//
#include <hip/hip_runtime.h>

typedef __attribute__((ext_vector_type(16))) __bf16 v16bf;
typedef __attribute__((ext_vector_type(8)))  __bf16 v8bf;
typedef __attribute__((ext_vector_type(8)))  float  v8f;

#define B_SZ   32
#define T_STEPS 128
#define I_CH   16
#define N_DIM  512

// ---------------------------------------------------------------------------
// Pass 1: repack W_g[i][n][k] (fp32) -> Wt[(g*16+i)*512 + k][n] (bf16).
// After this, a WMMA B-fragment lane (column k fixed, K=n contiguous) is two
// global b128 loads. Wt (33.5 MB) lives in d_ws and stays L2-resident.
// ---------------------------------------------------------------------------
__global__ void repack_w_kernel(const float* __restrict__ Wj,
                                const float* __restrict__ Wi,
                                const float* __restrict__ Wf,
                                const float* __restrict__ Wo,
                                __bf16* __restrict__ Wt) {
    const int g = blockIdx.y;
    const long e = (long)blockIdx.x * blockDim.x + threadIdx.x; // elem in gate
    const long per_gate = (long)I_CH * N_DIM * N_DIM;
    if (e >= per_gate) return;
    const int k = (int)(e % N_DIM);
    const int n = (int)((e / N_DIM) % N_DIM);
    const int i = (int)(e / ((long)N_DIM * N_DIM));
    const float* W = (g == 0) ? Wj : (g == 1) ? Wi : (g == 2) ? Wf : Wo;
    const float v = W[e];
    Wt[(((long)g * I_CH + i) * N_DIM + k) * N_DIM + n] = (__bf16)v;
}

// ---------------------------------------------------------------------------
// Load 16 bf16 from two 16B-aligned chunks into a WMMA fragment register set.
// ---------------------------------------------------------------------------
__device__ __forceinline__ v16bf load_frag(const __bf16* p0, const __bf16* p1) {
    v8bf lo = *(const v8bf*)p0;
    v8bf hi = *(const v8bf*)p1;
    v16bf r;
#pragma unroll
    for (int e = 0; e < 8; ++e) { r[e] = lo[e]; r[e + 8] = hi[e]; }
    return r;
}

__device__ __forceinline__ float fast_sigmoid(float x) {
    return 1.0f / (1.0f + __expf(-x));
}
__device__ __forceinline__ float fast_tanh(float x) {
    // clamp so exp() never overflows; tanh saturates long before 15
    x = fminf(fmaxf(x, -15.0f), 15.0f);
    float e = __expf(2.0f * x);
    return (e - 1.0f) / (e + 1.0f);
}

// ---------------------------------------------------------------------------
// Pass 2: persistent recurrence. One block per channel i (16 blocks).
// 512 threads = 16 wave32s. Wave w owns N-tiles {2w, 2w+1} x M-tiles {0,1};
// c/h state for those elements stays in VGPRs across all 128 timesteps.
// h is exchanged through 32 KB of LDS (bf16) with two barriers per step.
// ---------------------------------------------------------------------------
__global__ void __launch_bounds__(512, 1)
eslstm_kernel(const float* __restrict__ x,
              const float* __restrict__ Uj, const float* __restrict__ Ui,
              const float* __restrict__ Uf, const float* __restrict__ Uo,
              const float* __restrict__ bj, const float* __restrict__ bi,
              const float* __restrict__ bf_, const float* __restrict__ bo,
              const __bf16* __restrict__ Wt,
              float* __restrict__ out_seq,
              float* __restrict__ h_fin,
              float* __restrict__ c_fin) {
    __shared__ __bf16 h_lds[B_SZ * N_DIM]; // 32 KB of the 320 KB WGP LDS

    const int i_ch = blockIdx.x;
    const int tid  = threadIdx.x;
    const int wave = tid >> 5;
    const int lane = tid & 31;
    const int half = lane >> 4;   // which 16-lane half (WMMA layout selector)
    const int l15  = lane & 15;

    // h starts at zero
    for (int idx = tid; idx < B_SZ * N_DIM; idx += 512)
        h_lds[idx] = (__bf16)0.0f;
    __syncthreads();

    float c_reg[4][8];
    float h_reg[4][8];
#pragma unroll
    for (int p = 0; p < 4; ++p)
#pragma unroll
        for (int r = 0; r < 8; ++r) { c_reg[p][r] = 0.0f; h_reg[p][r] = 0.0f; }

    const float dtm   = -7.8125e-5f;   // -(DT/TAU_M)*0.5
    const float dta   = -1.5625e-4f;   // -(DT/TAU_ADP)*0.5
    const float a_out = __expf(-0.01f);
    const long  gstride = (long)I_CH * N_DIM * N_DIM;

#pragma unroll 1
    for (int t = 0; t < T_STEPS; ++t) {
#pragma unroll
        for (int pos = 0; pos < 4; ++pos) {
            const int mt    = pos >> 1;
            const int kt    = (wave << 1) | (pos & 1);
            const int col   = (kt << 4) + l15;          // output column
            const int mbase = (mt << 4) + (half << 3);  // first batch row

            // pre-activation init: x_t[b,i]*U[i,0,k] + b[i,k]
            const float Ujv = Uj[i_ch * N_DIM + col];
            const float Uiv = Ui[i_ch * N_DIM + col];
            const float Ufv = Uf[i_ch * N_DIM + col];
            const float Uov = Uo[i_ch * N_DIM + col];
            const float bjv = bj[i_ch * N_DIM + col];
            const float biv = bi[i_ch * N_DIM + col];
            const float bfv = bf_[i_ch * N_DIM + col];
            const float bov = bo[i_ch * N_DIM + col];

            v8f accj, acci, accf, acco;
#pragma unroll
            for (int r = 0; r < 8; ++r) {
                const float xv = x[((long)(mbase + r) * T_STEPS + t) * I_CH + i_ch];
                accj[r] = xv * Ujv + bjv;
                acci[r] = xv * Uiv + biv;
                accf[r] = xv * Ufv + bfv;
                acco[r] = xv * Uov + bov;
            }

            const __bf16* wj = Wt + (((long)0 * I_CH + i_ch) * N_DIM + col) * N_DIM;
            const __bf16* wi = wj + gstride;
            const __bf16* wf = wj + 2 * gstride;
            const __bf16* wo = wj + 3 * gstride;
            const __bf16* arow = h_lds + ((mt << 4) + l15) * N_DIM;

#pragma unroll 4
            for (int kk = 0; kk < 16; ++kk) {
                // A fragment (16x32 bf16): lane half selects K 0..15
                // vs 16..31 (contiguous, matching the B fragment).
                const int akoff = (kk << 5) + (half << 4);
                v16bf a = load_frag(arow + akoff, arow + akoff + 8);
                // B fragment (32x16 bf16): lane half selects K rows 0..15 vs
                // 16..31; 16 contiguous n-values thanks to the Wt repack.
                const int bkoff = (kk << 5) + (half << 4);
                v16bf bJ = load_frag(wj + bkoff, wj + bkoff + 8);
                accj = __builtin_amdgcn_wmma_f32_16x16x32_bf16(
                    false, a, false, bJ, (short)0, accj, false, false);
                v16bf bI = load_frag(wi + bkoff, wi + bkoff + 8);
                acci = __builtin_amdgcn_wmma_f32_16x16x32_bf16(
                    false, a, false, bI, (short)0, acci, false, false);
                v16bf bF = load_frag(wf + bkoff, wf + bkoff + 8);
                accf = __builtin_amdgcn_wmma_f32_16x16x32_bf16(
                    false, a, false, bF, (short)0, accf, false, false);
                v16bf bO = load_frag(wo + bkoff, wo + bkoff + 8);
                acco = __builtin_amdgcn_wmma_f32_16x16x32_bf16(
                    false, a, false, bO, (short)0, acco, false, false);
            }

            // spiking-neuron element-wise update on the accumulator fragment
#pragma unroll
            for (int r = 0; r < 8; ++r) {
                const float j  = fast_tanh(accj[r]);
                const float ig = fast_sigmoid(acci[r]);
                const float fg = fast_sigmoid(accf[r]);
                const float og = fast_sigmoid(acco[r]);
                const float h  = h_reg[pos][r];
                const float alpha_m = __expf(dtm * j);
                const float ro      = __expf(dta * ig);
                const float b_ad    = ro * 0.1f + (1.0f - ro) * ig;
                const float Bth     = 0.04f + 1.8f * b_ad;
                const float mem     = j * alpha_m + (1.0f - alpha_m) * h
                                      - Bth * ig * 0.01f;
                const float spike   = (mem - Bth) > 0.0f ? 1.0f : 0.0f;
                const float mem_out = mem * a_out + (1.0f - a_out) * spike + 0.08f;
                const float c_new   = c_reg[pos][r] * fg + ig * spike * mem_out;
                const float h_new   = og * fast_tanh(c_new);
                c_reg[pos][r] = c_new;
                h_reg[pos][r] = h_new;
                out_seq[(((long)(mbase + r) * T_STEPS + t) * I_CH + i_ch) * N_DIM
                        + col] = h_new;
            }
        } // pos

        __syncthreads(); // all h_lds reads of step t done
#pragma unroll
        for (int pos = 0; pos < 4; ++pos) {
            const int mt    = pos >> 1;
            const int kt    = (wave << 1) | (pos & 1);
            const int col   = (kt << 4) + l15;
            const int mbase = (mt << 4) + (half << 3);
#pragma unroll
            for (int r = 0; r < 8; ++r)
                h_lds[(mbase + r) * N_DIM + col] = (__bf16)h_reg[pos][r];
        }
        __syncthreads(); // h(t) visible to every wave for step t+1
    } // t

    // final h / c
#pragma unroll
    for (int pos = 0; pos < 4; ++pos) {
        const int mt    = pos >> 1;
        const int kt    = (wave << 1) | (pos & 1);
        const int col   = (kt << 4) + l15;
        const int mbase = (mt << 4) + (half << 3);
#pragma unroll
        for (int r = 0; r < 8; ++r) {
            const long o = ((long)(mbase + r) * I_CH + i_ch) * N_DIM + col;
            h_fin[o] = h_reg[pos][r];
            c_fin[o] = c_reg[pos][r];
        }
    }
}

extern "C" void kernel_launch(void* const* d_in, const int* in_sizes, int n_in,
                              void* d_out, int out_size, void* d_ws, size_t ws_size,
                              hipStream_t stream) {
    const float* x  = (const float*)d_in[0];
    const float* Uj = (const float*)d_in[1];
    const float* Ui = (const float*)d_in[2];
    const float* Uf = (const float*)d_in[3];
    const float* Uo = (const float*)d_in[4];
    const float* Wj = (const float*)d_in[5];
    const float* Wi = (const float*)d_in[6];
    const float* Wf = (const float*)d_in[7];
    const float* Wo = (const float*)d_in[8];
    const float* bj = (const float*)d_in[9];
    const float* bi = (const float*)d_in[10];
    const float* bf = (const float*)d_in[11];
    const float* bo = (const float*)d_in[12];

    __bf16* Wt = (__bf16*)d_ws; // 4*16*512*512 bf16 = 33.5 MB, L2-resident

    const long per_gate = (long)I_CH * N_DIM * N_DIM;
    dim3 rgrid((unsigned)((per_gate + 255) / 256), 4);
    repack_w_kernel<<<rgrid, 256, 0, stream>>>(Wj, Wi, Wf, Wo, Wt);

    float* out    = (float*)d_out;
    float* h_fin  = out + (long)B_SZ * T_STEPS * I_CH * N_DIM;
    float* c_fin  = h_fin + (long)B_SZ * I_CH * N_DIM;

    eslstm_kernel<<<dim3(I_CH), dim3(512), 0, stream>>>(
        x, Uj, Ui, Uf, Uo, bj, bi, bf, bo, Wt, out, h_fin, c_fin);
}
